// GATModel_154618823283
// MI455X (gfx1250) — matrix operand order repaired and checked by reading, not compiled, not run on hardware
//
#include <hip/hip_runtime.h>
#include <hip/hip_bf16.h>

typedef __attribute__((ext_vector_type(16))) _Float16 v16h;
typedef __attribute__((ext_vector_type(8)))  _Float16 v8h;
typedef __attribute__((ext_vector_type(8)))  float    v8f;

#define HID 64
#define NEG_SLOPE 0.2f
#define EPS_SM 1e-16f

// ---------------- helpers: monotone float<->uint key for atomic max ----------
__device__ __forceinline__ unsigned fenc(float f) {
  unsigned b = __float_as_uint(f);
  return b ^ (unsigned)(((int)b >> 31) | 0x80000000);
}
__device__ __forceinline__ float fdec(unsigned k) {
  unsigned b = (k & 0x80000000u) ? (k ^ 0x80000000u) : ~k;
  return __uint_as_float(b);
}
// fenc(-inf) == 0x007FFFFF
#define NEG_INF_KEY 0x007FFFFFu

// ---------------- generic utility kernels -----------------------------------
__global__ void k_zero_f32(float* __restrict__ p, long long n) {
  long long t = (long long)blockIdx.x * blockDim.x + threadIdx.x;
  if (t < n) p[t] = 0.0f;
}

__global__ void k_f32_to_f16(const float* __restrict__ s, _Float16* __restrict__ d, long long n) {
  long long t = (long long)blockIdx.x * blockDim.x + threadIdx.x;
  if (t < n) d[t] = (_Float16)s[t];
}

// ---------------- B pre-pack: W[K,64] f32 -> fragment-ordered f16 ------------
// Packed layout: Bp[(((kt*4 + tile)*32 + lane)*16) + j]
//   = W[(kt*32 + (lane>>4)*16 + j)*64 + tile*16 + (lane&15)]
// so each lane's 16-half B fragment is one contiguous 32-byte run.
__global__ void k_pack_B(const float* __restrict__ W, _Float16* __restrict__ Bp, int K) {
  int t = blockIdx.x * blockDim.x + threadIdx.x;
  if (t >= K * HID) return;
  int j    = t & 15;
  int lane = (t >> 4) & 31;
  int tile = (t >> 9) & 3;
  int kt   = t >> 11;
  int lrow = lane & 15;
  int hi   = lane >> 4;
  Bp[t] = (_Float16)W[(size_t)(kt * 32 + hi * 16 + j) * HID + tile * 16 + lrow];
}

// ---------------- WMMA GEMM: C[M,64] = A[M,K] * Bpacked ---------------------
// One wave32 computes a 16x64 tile of C.  K in {64,128}; M % 16 == 0.
template <int K>
__global__ void k_gemm_wmma(const _Float16* __restrict__ A,
                            const _Float16* __restrict__ Bp,
                            float* __restrict__ C, int M) {
  int wave = blockIdx.x * (blockDim.x >> 5) + (threadIdx.x >> 5);
  int m0 = wave * 16;
  if (m0 >= M) return;               // wave-uniform: EXEC all-ones at WMMA
  int lane = threadIdx.x & 31;
  int lrow = lane & 15;              // M (A) / N (B,C) sub-index
  int hi   = lane >> 4;              // lane-half select

  v8f acc0 = {}, acc1 = {}, acc2 = {}, acc3 = {};
  const _Float16* arow = A + (size_t)(m0 + lrow) * K + hi * 16;

#pragma unroll
  for (int kt = 0; kt < K / 32; ++kt) {
    // A fragment: two aligned 16B loads, concat into v16h
    const _Float16* ap = arow + kt * 32;
    v8h alo = *(const v8h*)(ap);
    v8h ahi = *(const v8h*)(ap + 8);
    v16h a = __builtin_shufflevector(alo, ahi,
        0, 1, 2, 3, 4, 5, 6, 7, 8, 9, 10, 11, 12, 13, 14, 15);

    // B fragments: contiguous 32B per lane from packed weights
    const _Float16* bbase = Bp + ((size_t)(kt * 4) * 32 + lane) * 16;
    v16h b0 = *(const v16h*)(bbase);
    v16h b1 = *(const v16h*)(bbase + 32 * 16);
    v16h b2 = *(const v16h*)(bbase + 2 * 32 * 16);
    v16h b3 = *(const v16h*)(bbase + 3 * 32 * 16);

    acc0 = __builtin_amdgcn_wmma_f32_16x16x32_f16(false, a, false, b0, (short)0, acc0, false, false);
    acc1 = __builtin_amdgcn_wmma_f32_16x16x32_f16(false, a, false, b1, (short)0, acc1, false, false);
    acc2 = __builtin_amdgcn_wmma_f32_16x16x32_f16(false, a, false, b2, (short)0, acc2, false, false);
    acc3 = __builtin_amdgcn_wmma_f32_16x16x32_f16(false, a, false, b3, (short)0, acc3, false, false);
  }

  // C tile store: VGPR r -> row m0 + r + hi*8, col = tile*16 + lrow
  float* cp = C + (size_t)(m0 + hi * 8) * HID + lrow;
#pragma unroll
  for (int r = 0; r < 8; ++r) {
    cp[(size_t)r * HID +  0] = acc0[r];
    cp[(size_t)r * HID + 16] = acc1[r];
    cp[(size_t)r * HID + 32] = acc2[r];
    cp[(size_t)r * HID + 48] = acc3[r];
  }
}

// ---------------- per-node: attention logits + softmax state init -----------
__global__ void k_node_prep(const float* __restrict__ h,
                            const float* __restrict__ a_s,
                            const float* __restrict__ a_d,
                            float* __restrict__ as, float* __restrict__ ad,
                            unsigned* __restrict__ mkey, float* __restrict__ ssum,
                            int n) {
  int i = blockIdx.x * blockDim.x + threadIdx.x;
  if (i >= n) return;
  const float4* hr = (const float4*)(h + (size_t)i * HID);
  float s = 0.f, d = 0.f;
#pragma unroll
  for (int c = 0; c < HID / 4; ++c) {
    float4 v = hr[c];
    s += v.x * a_s[4 * c + 0] + v.y * a_s[4 * c + 1] + v.z * a_s[4 * c + 2] + v.w * a_s[4 * c + 3];
    d += v.x * a_d[4 * c + 0] + v.y * a_d[4 * c + 1] + v.z * a_d[4 * c + 2] + v.w * a_d[4 * c + 3];
  }
  as[i] = s; ad[i] = d;
  mkey[i] = NEG_INF_KEY;
  ssum[i] = 0.f;
}

// ---------------- edge pass 1: leaky-relu logit + segment max ---------------
__global__ void k_edge_max(const int* __restrict__ ei, int E, int n,
                           const float* __restrict__ as, const float* __restrict__ ad,
                           float* __restrict__ ebuf, unsigned* __restrict__ mkey) {
  int k = blockIdx.x * blockDim.x + threadIdx.x;
  int ET = E + n;
  if (k >= ET) return;
  int s, d;
  if (k < E) { s = ei[k]; d = ei[E + k]; } else { s = d = k - E; }
  float e = as[s] + ad[d];
  e = (e > 0.f) ? e : NEG_SLOPE * e;
  ebuf[k] = e;
  atomicMax(&mkey[d], fenc(e));
}

// ---------------- edge pass 2: exp + segment sum ----------------------------
__global__ void k_edge_expsum(const int* __restrict__ ei, int E, int n,
                              float* __restrict__ ebuf,
                              const unsigned* __restrict__ mkey,
                              float* __restrict__ ssum) {
  int k = blockIdx.x * blockDim.x + threadIdx.x;
  int ET = E + n;
  if (k >= ET) return;
  int d = (k < E) ? ei[E + k] : (k - E);
  float w = __expf(ebuf[k] - fdec(mkey[d]));
  ebuf[k] = w;
  atomicAdd(&ssum[d], w);
}

// ---------------- edge pass 3: weighted scatter of messages -----------------
// one thread per (edge, channel): coalesced gather of h[src] row,
// coalesced atomic adds on out[dst] row (hot set resident in 192MB L2).
__global__ void k_edge_scatter(const int* __restrict__ ei, int E, int n,
                               const float* __restrict__ ebuf,
                               const float* __restrict__ ssum,
                               const float* __restrict__ h,
                               float* __restrict__ outacc) {
  long long t = (long long)blockIdx.x * blockDim.x + threadIdx.x;
  long long tot = (long long)(E + n) * HID;
  if (t >= tot) return;
  int k = (int)(t >> 6);
  int c = (int)(t & (HID - 1));
  int s, d;
  if (k < E) { s = ei[k]; d = ei[E + k]; } else { s = d = k - E; }
  float alpha = ebuf[k] / (ssum[d] + EPS_SM);
  atomicAdd(&outacc[(size_t)d * HID + c], alpha * h[(size_t)s * HID + c]);
}

// ---------------- per-(node,ch): bias + relu (+ f16 copy for next GEMM) -----
__global__ void k_node_finish(float* __restrict__ outacc,
                              const float* __restrict__ bias,
                              _Float16* __restrict__ x16, int n) {
  long long t = (long long)blockIdx.x * blockDim.x + threadIdx.x;
  if (t >= (long long)n * HID) return;
  int c = (int)(t & (HID - 1));
  float v = outacc[t] + bias[c];
  v = v > 0.f ? v : 0.f;
  outacc[t] = v;
  if (x16) x16[t] = (_Float16)v;
}

// ---------------- pooling ----------------------------------------------------
__global__ void k_pool_accum(const float* __restrict__ feat,
                             const int* __restrict__ batch,
                             float* __restrict__ pooled, float* __restrict__ cnt,
                             int n) {
  long long t = (long long)blockIdx.x * blockDim.x + threadIdx.x;
  if (t >= (long long)n * HID) return;
  int i = (int)(t >> 6);
  int c = (int)(t & (HID - 1));
  int g = batch[i];
  atomicAdd(&pooled[(size_t)g * HID + c], feat[t]);
  if (c == 0) atomicAdd(&cnt[g], 1.0f);
}

__global__ void k_pool_div(const float* __restrict__ pooled,
                           const float* __restrict__ cnt,
                           float* __restrict__ out, int ngraphs) {
  int t = blockIdx.x * blockDim.x + threadIdx.x;
  if (t >= ngraphs * HID) return;
  float c = cnt[t >> 6];
  out[t] = pooled[t] / (c > 1.0f ? c : 1.0f);
}

// ---------------- host-side orchestration -----------------------------------
extern "C" void kernel_launch(void* const* d_in, const int* in_sizes, int n_in,
                              void* d_out, int out_size, void* d_ws, size_t ws_size,
                              hipStream_t stream) {
  const float* x       = (const float*)d_in[0];
  const int*   ei      = (const int*)d_in[1];
  const int*   batch   = (const int*)d_in[2];
  const float* W1      = (const float*)d_in[3];
  const float* a_src1  = (const float*)d_in[4];
  const float* a_dst1  = (const float*)d_in[5];
  const float* b1      = (const float*)d_in[6];
  const float* W2      = (const float*)d_in[7];
  const float* a_src2  = (const float*)d_in[8];
  const float* a_dst2  = (const float*)d_in[9];
  const float* b2      = (const float*)d_in[10];

  const int N    = in_sizes[2];          // 100000
  const int E    = in_sizes[1] / 2;      // 1600000
  const int INCH = in_sizes[0] / N;      // 128
  const int ET   = E + N;                // with self-loops
  const int NG   = out_size / HID;       // 64 graphs

  // ---- workspace carve-up (256B aligned) ----
  char* ws = (char*)d_ws;
  size_t off = 0;
  auto carve = [&](size_t bytes) -> char* {
    char* p = ws + off;
    off = (off + bytes + 255) & ~(size_t)255;
    return p;
  };
  float*     H     = (float*)    carve((size_t)N * HID * sizeof(float));      // current h
  float*     OUT   = (float*)    carve((size_t)N * HID * sizeof(float));      // accumulator
  _Float16*  X16   = (_Float16*) carve((size_t)N * INCH * sizeof(_Float16));  // f16 activations
  _Float16*  W16a  = (_Float16*) carve((size_t)INCH * HID * sizeof(_Float16)); // packed
  _Float16*  W16b  = (_Float16*) carve((size_t)HID * HID * sizeof(_Float16));  // packed
  float*     AS    = (float*)    carve((size_t)N * sizeof(float));
  float*     AD    = (float*)    carve((size_t)N * sizeof(float));
  unsigned*  MKEY  = (unsigned*) carve((size_t)N * sizeof(unsigned));
  float*     SSUM  = (float*)    carve((size_t)N * sizeof(float));
  float*     EBUF  = (float*)    carve((size_t)ET * sizeof(float));
  float*     POOL  = (float*)    carve((size_t)NG * HID * sizeof(float));
  float*     CNT   = (float*)    carve((size_t)NG * sizeof(float));

  const int TB = 256;
  auto blks   = [&](long long n) { return (int)((n + TB - 1) / TB); };

  // ---- precision staging: activations f32->f16, weights f32->f16 + pack ----
  k_f32_to_f16<<<blks((long long)N * INCH), TB, 0, stream>>>(x, X16, (long long)N * INCH);
  k_pack_B<<<blks((long long)INCH * HID), TB, 0, stream>>>(W1, W16a, INCH);
  k_pack_B<<<blks((long long)HID * HID),  TB, 0, stream>>>(W2, W16b, HID);

  const int gemmBlocks = (N / 16 + 7) / 8;   // 8 waves per 256-thread block

  // =================== layer 1 ===================
  if (INCH == 128)     k_gemm_wmma<128><<<gemmBlocks, TB, 0, stream>>>(X16, W16a, H, N);
  else /* INCH==64 */  k_gemm_wmma< 64><<<gemmBlocks, TB, 0, stream>>>(X16, W16a, H, N);
  k_node_prep<<<blks(N), TB, 0, stream>>>(H, a_src1, a_dst1, AS, AD, MKEY, SSUM, N);
  k_zero_f32<<<blks((long long)N * HID), TB, 0, stream>>>(OUT, (long long)N * HID);
  k_edge_max<<<blks(ET), TB, 0, stream>>>(ei, E, N, AS, AD, EBUF, MKEY);
  k_edge_expsum<<<blks(ET), TB, 0, stream>>>(ei, E, N, EBUF, MKEY, SSUM);
  k_edge_scatter<<<blks((long long)ET * HID), TB, 0, stream>>>(ei, E, N, EBUF, SSUM, H, OUT);
  k_node_finish<<<blks((long long)N * HID), TB, 0, stream>>>(OUT, b1, X16, N); // relu -> X16 (f16)

  // =================== layer 2 ===================
  k_gemm_wmma<64><<<gemmBlocks, TB, 0, stream>>>(X16, W16b, H, N);
  k_node_prep<<<blks(N), TB, 0, stream>>>(H, a_src2, a_dst2, AS, AD, MKEY, SSUM, N);
  k_zero_f32<<<blks((long long)N * HID), TB, 0, stream>>>(OUT, (long long)N * HID);
  k_edge_max<<<blks(ET), TB, 0, stream>>>(ei, E, N, AS, AD, EBUF, MKEY);
  k_edge_expsum<<<blks(ET), TB, 0, stream>>>(ei, E, N, EBUF, MKEY, SSUM);
  k_edge_scatter<<<blks((long long)ET * HID), TB, 0, stream>>>(ei, E, N, EBUF, SSUM, H, OUT);
  k_node_finish<<<blks((long long)N * HID), TB, 0, stream>>>(OUT, b2, (_Float16*)nullptr, N);

  // =================== global mean pool ===================
  k_zero_f32<<<blks((long long)NG * HID), TB, 0, stream>>>(POOL, (long long)NG * HID);
  k_zero_f32<<<1, TB, 0, stream>>>(CNT, NG);
  k_pool_accum<<<blks((long long)N * HID), TB, 0, stream>>>(OUT, batch, POOL, CNT, N);
  k_pool_div<<<blks(NG * HID), TB, 0, stream>>>(POOL, CNT, (float*)d_out, NG);
}
